// IASA_79654463471806
// MI455X (gfx1250) — matrix operand order repaired and checked by reading, not compiled, not run on hardware
//
#include <hip/hip_runtime.h>
#include <hip/hip_bf16.h>

typedef __bf16 bf16;
typedef __attribute__((ext_vector_type(16))) __bf16 v16bf;
typedef __attribute__((ext_vector_type(8)))  __bf16 v8bf;
typedef __attribute__((ext_vector_type(8)))  float  v8f;
typedef __attribute__((ext_vector_type(4)))  float  v4f;

#define WMMA_BF16(a,b,c) __builtin_amdgcn_wmma_f32_16x16x32_bf16(false,(a),false,(b),(short)0,(c),false,false)

// Problem constants
#define Bx    2
#define Nn    16200
#define Mm    128
#define DIMc  256
#define Hh    8
#define DHd   32
#define GSc   128
#define NGc   127      // ceil(16200/128)
#define RPAD  16384    // (NG+1)*GS padded row count (covers q pad 16256 and k/v pad 16384)

__device__ __forceinline__ v8f zero_v8f() {
    v8f z; 
#pragma unroll
    for (int i = 0; i < 8; i++) z[i] = 0.f;
    return z;
}

__device__ __forceinline__ v16bf make_a_from_halves(v8bf lo, v8bf hi) {
    v16bf a;
#pragma unroll
    for (int i = 0; i < 8; i++) { a[i] = lo[i]; a[8 + i] = hi[i]; }
    return a;
}

// ---------------------------------------------------------------------------
// Kernel 0: transpose + convert six 256x256 fp32 weights -> bf16 Wt[n][k]
// ---------------------------------------------------------------------------
__global__ void iasa_wt_prep(const float* w0, const float* w1, const float* w2,
                             const float* w3, const float* w4, const float* w5,
                             bf16* wt) {
    const float* ws[6] = {w0, w1, w2, w3, w4, w5};
    const float* W = ws[blockIdx.x];
    bf16* dst = wt + (size_t)blockIdx.x * 65536;
    for (int i = threadIdx.x; i < 65536; i += blockDim.x) {
        int n = i >> 8, k = i & 255;
        dst[(n << 8) + k] = (bf16)W[(k << 8) + n];
    }
}

// ---------------------------------------------------------------------------
// Kernel 1: projection GEMM  Out[r] = bf16( X[src(r)] @ W )
//   wrap=1: src = r<srcN ? r : 2*srcN-1-r   (reference flip-tail padding)
//   tout=1: store transposed Out[col][row] with leading dim ldt
// Block 256 threads = 8 waves; tile 128 rows x 64 cols; K=256 (8 wmma steps)
// ---------------------------------------------------------------------------
__global__ void iasa_proj_gemm(const float* __restrict__ X, int xbs, int srcN, int wrap,
                               const bf16* __restrict__ Wt, bf16* __restrict__ Out,
                               long obs, int tout, int ldt) {
    const int b    = blockIdx.z;
    const int lane = threadIdx.x & 31;
    const int w    = threadIdx.x >> 5;
    const int r0   = blockIdx.x * 128 + w * 16;
    const int c0   = blockIdx.y * 64;
    const int half = lane >> 4;            // 0 or 1
    const int hs8  = half << 3;            // unused (0 / 8); A uses hs16
    const int hs16 = half << 4;            // B chunk select (0 / 16)

    int arow = r0 + (lane & 15);
    int src  = wrap ? (arow < srcN ? arow : 2 * srcN - 1 - arow)
                    : (arow < srcN ? arow : srcN - 1);
    const float* xr = X + (size_t)b * xbs + (size_t)src * DIMc;

    v8f acc[4];
#pragma unroll
    for (int i = 0; i < 4; i++) acc[i] = zero_v8f();

#pragma unroll
    for (int s = 0; s < 8; s++) {
        const int k0 = s * 32;
        // A fragment: lane<16 holds K {k0..k0+15}; lane>=16 shifted by 16
        v4f f0 = *(const v4f*)(xr + k0 + hs16);
        v4f f1 = *(const v4f*)(xr + k0 + hs16 + 4);
        v4f f2 = *(const v4f*)(xr + k0 + hs16 + 8);
        v4f f3 = *(const v4f*)(xr + k0 + hs16 + 12);
        v16bf a;
#pragma unroll
        for (int i = 0; i < 4; i++) {
            a[i]      = (bf16)f0[i];
            a[4 + i]  = (bf16)f1[i];
            a[8 + i]  = (bf16)f2[i];
            a[12 + i] = (bf16)f3[i];
        }
#pragma unroll
        for (int c = 0; c < 4; c++) {
            const int col = c0 + c * 16 + (lane & 15);
            v16bf bb = *(const v16bf*)(Wt + (size_t)col * DIMc + k0 + hs16);
            acc[c] = WMMA_BF16(a, bb, acc[c]);
        }
    }
    // D layout: VGPR j -> row j + 8*half; col = lane&15
#pragma unroll
    for (int c = 0; c < 4; c++) {
        const int col = c0 + c * 16 + (lane & 15);
#pragma unroll
        for (int j = 0; j < 8; j++) {
            const int rr = r0 + j + (half << 3);
            bf16 v = (bf16)acc[c][j];
            if (tout) Out[(size_t)b * obs + (size_t)col * ldt + rr] = v;
            else      Out[(size_t)b * obs + (size_t)rr * DIMc + col] = v;
        }
    }
}

// ---------------------------------------------------------------------------
// Kernel 2: attention. grid = (NG, H, B), block = 256 (8 waves).
// Each wave: 16 queries. Local window (256 keys) + prototype (128 keys),
// separate softmaxes, o1+o2 accumulated in the same WMMA C registers,
// scatter-stored via idx permutation.
// Dynamic LDS: 8 waves * 16 * 256 bf16 = 64 KB.
// ---------------------------------------------------------------------------
__global__ void iasa_attn(const bf16* __restrict__ qb, const bf16* __restrict__ kb,
                          const bf16* __restrict__ vt, const bf16* __restrict__ kg,
                          const bf16* __restrict__ vgt, const int* __restrict__ idx,
                          bf16* __restrict__ outs) {
    extern __shared__ char smem_raw[];
    bf16* smem = (bf16*)smem_raw;

    const int g = blockIdx.x, h = blockIdx.y, b = blockIdx.z;
    const int lane = threadIdx.x & 31;
    const int w    = threadIdx.x >> 5;
    const int half = lane >> 4;
    const int hs8  = half << 3;
    const int hs16 = half << 4;
    bf16* sp = smem + (size_t)w * 16 * 256;

    const float scale = 0.1767766952966369f; // 1/sqrt(32)
    const int qrow0 = g * GSc + w * 16;

    // ---- Q A-fragment (dh = 32 = one WMMA K-step), loaded once ----
    const int qr = qrow0 + (lane & 15);
    const bf16* qp = qb + ((size_t)b * RPAD + qr) * DIMc + h * DHd;
    v16bf qa = make_a_from_halves(*(const v8bf*)(qp + hs16),
                                  *(const v8bf*)(qp + hs16 + 8));

    // ---- local scores S1 = Q @ Kw^T  (window = rows [g*128, g*128+256)) ----
#pragma unroll
    for (int t = 0; t < 16; t++) {
        const int key = g * GSc + t * 16 + (lane & 15);
        v16bf kbf = *(const v16bf*)(kb + ((size_t)b * RPAD + key) * DIMc + h * DHd + hs16);
        v8f s = WMMA_BF16(qa, kbf, zero_v8f());
#pragma unroll
        for (int j = 0; j < 8; j++) {
            const int m = j + (half << 3);
            sp[m * 256 + t * 16 + (lane & 15)] = (bf16)(s[j] * scale);
        }
    }
    __syncthreads();

    // ---- softmax over 256 local keys (2 lanes per row, combine via shfl) ----
    {
        const int row = lane & 15;
        const int c0 = half * 128;
        float mx = -3.0e38f;
        for (int c = 0; c < 128; c++) mx = fmaxf(mx, (float)sp[row * 256 + c0 + c]);
        mx = fmaxf(mx, __shfl_xor(mx, 16, 32));
        float sum = 0.f;
        for (int c = 0; c < 128; c++) sum += __expf((float)sp[row * 256 + c0 + c] - mx);
        sum += __shfl_xor(sum, 16, 32);
        const float rinv = 1.f / sum;
        for (int c = 0; c < 128; c++) {
            float v = (float)sp[row * 256 + c0 + c];
            sp[row * 256 + c0 + c] = (bf16)(__expf(v - mx) * rinv);
        }
    }
    __syncthreads();

    // ---- O1 = P1 @ Vw  (K loop 8x32, B frags contiguous from transposed V) ----
    v8f acc[2];
    acc[0] = zero_v8f(); acc[1] = zero_v8f();
#pragma unroll
    for (int s8 = 0; s8 < 8; s8++) {
        const int k0 = s8 * 32;
        const bf16* pp = sp + (lane & 15) * 256 + k0 + hs16;
        v16bf pa = make_a_from_halves(*(const v8bf*)pp, *(const v8bf*)(pp + 8));
#pragma unroll
        for (int ct = 0; ct < 2; ct++) {
            const int dim = h * DHd + ct * 16 + (lane & 15);
            v16bf vb = *(const v16bf*)(vt + ((size_t)b * DIMc + dim) * RPAD
                                       + g * GSc + k0 + hs16);
            acc[ct] = WMMA_BF16(pa, vb, acc[ct]);
        }
    }
    __syncthreads();

    // ---- global scores S2 = Q @ Kg^T (128 prototypes) ----
#pragma unroll
    for (int t = 0; t < 8; t++) {
        const int proto = t * 16 + (lane & 15);
        v16bf kbf = *(const v16bf*)(kg + ((size_t)b * Mm + proto) * DIMc + h * DHd + hs16);
        v8f s = WMMA_BF16(qa, kbf, zero_v8f());
#pragma unroll
        for (int j = 0; j < 8; j++) {
            const int m = j + (half << 3);
            sp[m * 256 + t * 16 + (lane & 15)] = (bf16)(s[j] * scale);
        }
    }
    __syncthreads();

    // ---- softmax over 128 prototypes ----
    {
        const int row = lane & 15;
        const int c0 = half * 64;
        float mx = -3.0e38f;
        for (int c = 0; c < 64; c++) mx = fmaxf(mx, (float)sp[row * 256 + c0 + c]);
        mx = fmaxf(mx, __shfl_xor(mx, 16, 32));
        float sum = 0.f;
        for (int c = 0; c < 64; c++) sum += __expf((float)sp[row * 256 + c0 + c] - mx);
        sum += __shfl_xor(sum, 16, 32);
        const float rinv = 1.f / sum;
        for (int c = 0; c < 64; c++) {
            float v = (float)sp[row * 256 + c0 + c];
            sp[row * 256 + c0 + c] = (bf16)(__expf(v - mx) * rinv);
        }
    }
    __syncthreads();

    // ---- O += P2 @ Vg (same accumulators: o1+o2 is linear) ----
#pragma unroll
    for (int s8 = 0; s8 < 4; s8++) {
        const int k0 = s8 * 32;
        const bf16* pp = sp + (lane & 15) * 256 + k0 + hs16;
        v16bf pa = make_a_from_halves(*(const v8bf*)pp, *(const v8bf*)(pp + 8));
#pragma unroll
        for (int ct = 0; ct < 2; ct++) {
            const int dim = h * DHd + ct * 16 + (lane & 15);
            v16bf vb = *(const v16bf*)(vgt + ((size_t)b * DIMc + dim) * Mm + k0 + hs16);
            acc[ct] = WMMA_BF16(pa, vb, acc[ct]);
        }
    }

    // ---- scatter store through idx permutation (fused scatter-back) ----
#pragma unroll
    for (int ct = 0; ct < 2; ct++) {
        const int dcol = h * DHd + ct * 16 + (lane & 15);
#pragma unroll
        for (int j = 0; j < 8; j++) {
            const int m = j + (half << 3);
            const int grow = qrow0 + m;
            if (grow < Nn) {
                const int dest = idx[(size_t)b * Nn + grow];
                outs[((size_t)b * RPAD + dest) * DIMc + dcol] = (bf16)acc[ct][j];
            }
        }
    }
}

// ---------------------------------------------------------------------------
// Kernel 3: final projection  d_out = outs(bf16) @ Wproj, fp32 output, row guard
// ---------------------------------------------------------------------------
__global__ void iasa_final_gemm(const bf16* __restrict__ A, const bf16* __restrict__ Wt,
                                float* __restrict__ Out) {
    const int b    = blockIdx.z;
    const int lane = threadIdx.x & 31;
    const int w    = threadIdx.x >> 5;
    const int r0   = blockIdx.x * 128 + w * 16;
    const int c0   = blockIdx.y * 64;
    const int half = lane >> 4;
    const int hs8  = half << 3;
    const int hs16 = half << 4;

    const bf16* ar = A + ((size_t)b * RPAD + r0 + (lane & 15)) * DIMc;

    v8f acc[4];
#pragma unroll
    for (int i = 0; i < 4; i++) acc[i] = zero_v8f();

#pragma unroll
    for (int s = 0; s < 8; s++) {
        const int k0 = s * 32;
        v16bf a = make_a_from_halves(*(const v8bf*)(ar + k0 + hs16),
                                     *(const v8bf*)(ar + k0 + hs16 + 8));
#pragma unroll
        for (int c = 0; c < 4; c++) {
            const int col = c0 + c * 16 + (lane & 15);
            v16bf bb = *(const v16bf*)(Wt + (size_t)col * DIMc + k0 + hs16);
            acc[c] = WMMA_BF16(a, bb, acc[c]);
        }
    }
#pragma unroll
    for (int c = 0; c < 4; c++) {
        const int col = c0 + c * 16 + (lane & 15);
#pragma unroll
        for (int j = 0; j < 8; j++) {
            const int rr = r0 + j + (half << 3);
            if (rr < Nn) Out[((size_t)b * Nn + rr) * DIMc + col] = acc[c][j];
        }
    }
}

// ---------------------------------------------------------------------------
// Launch
// ---------------------------------------------------------------------------
extern "C" void kernel_launch(void* const* d_in, const int* in_sizes, int n_in,
                              void* d_out, int out_size, void* d_ws, size_t ws_size,
                              hipStream_t stream) {
    (void)in_sizes; (void)n_in; (void)out_size; (void)ws_size;

    const float* x      = (const float*)d_in[0];   // sorted_x  [B,N,256]
    const int*   idx    = (const int*)d_in[1];     // idx_last  [B,N,1]
    const float* protos = (const float*)d_in[2];   // prototypes[B,128,256]
    const float* Wq     = (const float*)d_in[3];
    const float* Wk     = (const float*)d_in[4];
    const float* Wv     = (const float*)d_in[5];
    const float* Wpk    = (const float*)d_in[6];
    const float* Wpv    = (const float*)d_in[7];
    const float* Wproj  = (const float*)d_in[8];
    float* out = (float*)d_out;

    // Workspace layout (bf16 elements)
    bf16* ws = (bf16*)d_ws;
    size_t o = 0;
    bf16* wt   = ws + o; o += (size_t)6 * 65536;          // 6 transposed weights
    bf16* qb   = ws + o; o += (size_t)Bx * RPAD * DIMc;   // q  (padded, natural)
    bf16* kb   = ws + o; o += (size_t)Bx * RPAD * DIMc;   // k  (padded, natural)
    bf16* vtb  = ws + o; o += (size_t)Bx * DIMc * RPAD;   // v  (padded, transposed [dim][key])
    bf16* kgb  = ws + o; o += (size_t)Bx * Mm * DIMc;     // kg (natural)
    bf16* vgt  = ws + o; o += (size_t)Bx * DIMc * Mm;     // vg (transposed [dim][proto])
    bf16* outs = ws + o; o += (size_t)Bx * RPAD * DIMc;   // scattered attention output

    dim3 blk(256);

    // 0) weight transpose+convert
    iasa_wt_prep<<<dim3(6), blk, 0, stream>>>(Wq, Wk, Wv, Wpk, Wpv, Wproj, wt);

    // 1) projections (q,k natural; v transposed; kg natural; vg transposed)
    iasa_proj_gemm<<<dim3(RPAD / 128, 4, Bx), blk, 0, stream>>>(
        x, Nn * DIMc, Nn, 1, wt + 0 * 65536, qb, (long)RPAD * DIMc, 0, 0);
    iasa_proj_gemm<<<dim3(RPAD / 128, 4, Bx), blk, 0, stream>>>(
        x, Nn * DIMc, Nn, 1, wt + 1 * 65536, kb, (long)RPAD * DIMc, 0, 0);
    iasa_proj_gemm<<<dim3(RPAD / 128, 4, Bx), blk, 0, stream>>>(
        x, Nn * DIMc, Nn, 1, wt + 2 * 65536, vtb, (long)DIMc * RPAD, 1, RPAD);
    iasa_proj_gemm<<<dim3(1, 4, Bx), blk, 0, stream>>>(
        protos, Mm * DIMc, Mm, 0, wt + 3 * 65536, kgb, (long)Mm * DIMc, 0, 0);
    iasa_proj_gemm<<<dim3(1, 4, Bx), blk, 0, stream>>>(
        protos, Mm * DIMc, Mm, 0, wt + 4 * 65536, vgt, (long)DIMc * Mm, 1, Mm);

    // 2) attention (local window + prototypes), scatter fused
    iasa_attn<<<dim3(NGc, Hh, Bx), blk, 8 * 16 * 256 * sizeof(bf16), stream>>>(
        qb, kb, vtb, kgb, vgt, idx, outs);

    // 3) final projection -> fp32 d_out
    iasa_final_gemm<<<dim3(NGc, 4, Bx), blk, 0, stream>>>(outs, wt + 5 * 65536, out);
}
